// VASNet_56702158241878
// MI455X (gfx1250) — compile-verified, matrix operand order checked by reading, not run on hardware
//
#include <hip/hip_runtime.h>
#include <hip/hip_bf16.h>

// ---------------------------------------------------------------------------
// VASNet block for MI455X (gfx1250, wave32). All GEMMs run on
// v_wmma_f32_16x16x32_bf16 with operands pre-staged in bf16 so the inner
// loops are pure {global_load_b128 x16, v_wmma x16} per 32-deep k-step.
// ---------------------------------------------------------------------------
#define NR 8192     // sequence length (rows)
#define MD 1024     // model dim

typedef __attribute__((ext_vector_type(16))) __bf16 v16bf;
typedef __attribute__((ext_vector_type(8)))  __bf16 v8bf;
typedef __attribute__((ext_vector_type(8)))  float  v8f;

// f32 -> bf16 round-to-nearest-even (used only in O(N) staging/epilogues)
static __device__ inline __bf16 f2bf(float f) {
  unsigned u = __builtin_bit_cast(unsigned, f);
  u += 0x7FFFu + ((u >> 16) & 1u);
  unsigned short h = (unsigned short)(u >> 16);
  return __builtin_bit_cast(__bf16, h);
}
static __device__ inline __bf16 bf_zero() {
  return __builtin_bit_cast(__bf16, (unsigned short)0);
}

static __device__ inline v8f wmma_bf16(v16bf a, v16bf b, v8f c) {
  return __builtin_amdgcn_wmma_f32_16x16x32_bf16(
      /*neg_a=*/false, a, /*neg_b=*/false, b,
      /*c_mod=*/(short)0, c, /*reuse_a=*/false, /*reuse_b=*/false);
}

// 16x32 bf16 A/B fragment from row-major bf16, K-contiguous.
// Lane l: row = r0 + (l&15); k-base = k0 + 8*(l>>4);
// elems [0..7] = k..k+7, [8..15] = k+16..k+23  (CDNA5 16-bit 16x32 layout).
// Two 16B contiguous loads per lane -> 2x global_load_b128, no conversion.
static __device__ inline v16bf load_frag_bf(const __bf16* __restrict__ src,
                                            int ld, int r0, int k0) {
  int lane = threadIdx.x & 31;
  const __bf16* p = src + (size_t)(r0 + (lane & 15)) * ld + (k0 + 8 * (lane >> 4));
  v8bf lo = *(const v8bf*)p;
  v8bf hi = *(const v8bf*)(p + 16);
  return __builtin_shufflevector(lo, hi, 0, 1, 2, 3, 4, 5, 6, 7,
                                 8, 9, 10, 11, 12, 13, 14, 15);
}

// ---------------------------------------------------------------------------
// Generic bf16 GEMM: C = act(scale * A @ B^T + bias + residual)
// A: M x Kd (row major bf16), B: N x Kd (row major bf16, weight layout).
// Outputs (any subset): Cf (f32 row major, ld=ldc), Cb (bf16 row major,
// ld=ldc), Ct (bf16 TRANSPOSED, Ct[n*ldt + m]).
// Workgroup = 8 waves; wave tile 64x64 (acc[4][4] v8f = 128 VGPRs);
// workgroup tile 128x256. Per k-step: 16 b128 loads feed 16 WMMAs.
// ---------------------------------------------------------------------------
__global__ __launch_bounds__(256) void gemm_bf16_kernel(
    const __bf16* __restrict__ A, const __bf16* __restrict__ B,
    int Kd, int ldc,
    float* __restrict__ Cf, __bf16* __restrict__ Cb,
    __bf16* __restrict__ Ct, int ldt,
    float scale, const float* __restrict__ bias,
    const float* __restrict__ residual, int do_relu) {
  int wave = threadIdx.x >> 5;
  int row0 = blockIdx.y * 128 + (wave >> 2) * 64;
  int col0 = blockIdx.x * 256 + (wave & 3) * 64;
  v8f acc[4][4] = {};
  for (int k0 = 0; k0 < Kd; k0 += 32) {
    v16bf a[4];
#pragma unroll
    for (int rt = 0; rt < 4; ++rt) a[rt] = load_frag_bf(A, Kd, row0 + rt * 16, k0);
#pragma unroll
    for (int ct = 0; ct < 4; ++ct) {
      v16bf b = load_frag_bf(B, Kd, col0 + ct * 16, k0);
#pragma unroll
      for (int rt = 0; rt < 4; ++rt)
        acc[rt][ct] = wmma_bf16(a[rt], b, acc[rt][ct]);
    }
  }
  int lane = threadIdx.x & 31;
  int nlo = lane & 15, mb8 = 8 * (lane >> 4);
#pragma unroll
  for (int rt = 0; rt < 4; ++rt) {
#pragma unroll
    for (int ct = 0; ct < 4; ++ct) {
      int n = col0 + ct * 16 + nlo;
      int rb = row0 + rt * 16 + mb8;
      float bv = bias ? bias[n] : 0.f;
#pragma unroll
      for (int v = 0; v < 8; ++v) {
        int r = rb + v;
        float val = acc[rt][ct][v] * scale + bv;
        if (residual) val += residual[(size_t)r * ldc + n];
        if (do_relu)  val = fmaxf(val, 0.f);
        if (Cf) Cf[(size_t)r * ldc + n] = val;
        if (Cb) Cb[(size_t)r * ldc + n] = f2bf(val);
        if (Ct) Ct[(size_t)n * ldt + r] = f2bf(val);
      }
    }
  }
}

// ---------------------------------------------------------------------------
// mn GEMM: MN = sum_d relu(Dd @ W^T + bias), D1/D2/D3 pre-staged bf16 diffs.
// Wave tile 16x64 with 3 accumulator sets sharing each B fragment
// (acc[3][4] = 96 VGPRs); per k-step: 7 fragments -> 12 WMMAs.
// ---------------------------------------------------------------------------
__global__ __launch_bounds__(256) void mn_gemm_kernel(
    const __bf16* __restrict__ D1, const __bf16* __restrict__ D2,
    const __bf16* __restrict__ D3, const __bf16* __restrict__ W,
    const float* __restrict__ bias, float* __restrict__ MN) {
  int wave = threadIdx.x >> 5;
  int row0 = blockIdx.y * 64 + (wave >> 1) * 16;
  int col0 = blockIdx.x * 128 + (wave & 1) * 64;
  v8f acc[3][4] = {};
  for (int k0 = 0; k0 < MD; k0 += 32) {
    v16bf a1 = load_frag_bf(D1, MD, row0, k0);
    v16bf a2 = load_frag_bf(D2, MD, row0, k0);
    v16bf a3 = load_frag_bf(D3, MD, row0, k0);
#pragma unroll
    for (int ct = 0; ct < 4; ++ct) {
      v16bf b = load_frag_bf(W, MD, col0 + ct * 16, k0);
      acc[0][ct] = wmma_bf16(a1, b, acc[0][ct]);
      acc[1][ct] = wmma_bf16(a2, b, acc[1][ct]);
      acc[2][ct] = wmma_bf16(a3, b, acc[2][ct]);
    }
  }
  int lane = threadIdx.x & 31;
  int nlo = lane & 15, mb8 = 8 * (lane >> 4);
#pragma unroll
  for (int ct = 0; ct < 4; ++ct) {
    int n = col0 + ct * 16 + nlo;
    float bv = bias[n];
#pragma unroll
    for (int v = 0; v < 8; ++v) {
      int r = row0 + mb8 + v;
      if (r < NR - 4) {
        float s = fmaxf(acc[0][ct][v] + bv, 0.f) +
                  fmaxf(acc[1][ct][v] + bv, 0.f) +
                  fmaxf(acc[2][ct][v] + bv, 0.f);
        MN[(size_t)r * MD + n] = s;
      }
    }
  }
}

__global__ void mn_tail_kernel(const float* __restrict__ X, float* __restrict__ MN) {
  int i = blockIdx.x * blockDim.x + threadIdx.x;
  if (i < 4 * MD) {
    size_t off = (size_t)(NR - 4) * MD + i;
    MN[off] = X[off];
  }
}

// ---------------------------------------------------------------------------
// Staging kernels (one-shot conversions; streaming, negligible vs GEMMs)
// ---------------------------------------------------------------------------
__global__ void cvt_bf16_kernel(const float* __restrict__ src,
                                __bf16* __restrict__ dst, int count) {
  int i = blockIdx.x * blockDim.x + threadIdx.x;
  if (i < count) dst[i] = f2bf(src[i]);
}

// d{1,2,4}[r][c] = bf16(|x[r+d][c] - x[r][c]|) for r < NR-4; 0 elsewhere.
__global__ void diff_kernel(const float* __restrict__ X, __bf16* __restrict__ d1,
                            __bf16* __restrict__ d2, __bf16* __restrict__ d3) {
  size_t i = (size_t)blockIdx.x * blockDim.x + threadIdx.x;
  size_t r = i >> 10;  // / MD
  float b = X[i];
  if (r < NR - 4) {
    d1[i] = f2bf(fabsf(X[i + MD] - b));
    d2[i] = f2bf(fabsf(X[i + 2 * MD] - b));
    d3[i] = f2bf(fabsf(X[i + 4 * MD] - b));
  } else {
    d1[i] = bf_zero(); d2[i] = bf_zero(); d3[i] = bf_zero();
  }
}

// awT[i*NR + j] = bf16(aw[j*NR + i]) via 64x64 LDS tile (coalesced both ways)
__global__ __launch_bounds__(256) void transpose_bf_kernel(
    const float* __restrict__ A, __bf16* __restrict__ At) {
  __shared__ __bf16 tile[64][65];
  int bx = blockIdx.x * 64;  // col offset in A
  int by = blockIdx.y * 64;  // row offset in A
  int tx = threadIdx.x & 63;
  int ty = threadIdx.x >> 6;  // 0..3
#pragma unroll
  for (int i = 0; i < 16; ++i) {
    int row = ty * 16 + i;
    tile[row][tx] = f2bf(A[(size_t)(by + row) * NR + bx + tx]);
  }
  __syncthreads();
#pragma unroll
  for (int i = 0; i < 16; ++i) {
    int row = ty * 16 + i;  // row in At = col in A
    At[(size_t)(bx + row) * NR + by + tx] = tile[tx][row];
  }
}

// ---------------------------------------------------------------------------
// In-place row softmax over NR columns; one workgroup/row, 32 vals/thread.
// ---------------------------------------------------------------------------
__global__ __launch_bounds__(256) void softmax_kernel(float* __restrict__ aw) {
  __shared__ float red[256];
  int tid = threadIdx.x;
  float* p = aw + (size_t)blockIdx.x * NR;
  float v[32];
  float mx = -3.4e38f;
#pragma unroll
  for (int i = 0; i < 32; ++i) { v[i] = p[i * 256 + tid]; mx = fmaxf(mx, v[i]); }
  red[tid] = mx; __syncthreads();
  for (int s = 128; s > 0; s >>= 1) {
    if (tid < s) red[tid] = fmaxf(red[tid], red[tid + s]);
    __syncthreads();
  }
  mx = red[0]; __syncthreads();
  float sum = 0.f;
#pragma unroll
  for (int i = 0; i < 32; ++i) { v[i] = __expf(v[i] - mx); sum += v[i]; }
  red[tid] = sum; __syncthreads();
  for (int s = 128; s > 0; s >>= 1) {
    if (tid < s) red[tid] += red[tid + s];
    __syncthreads();
  }
  float inv = 1.f / red[0];
#pragma unroll
  for (int i = 0; i < 32; ++i) p[i * 256 + tid] = v[i] * inv;
}

// ---------------------------------------------------------------------------
// Row LayerNorm over MD columns. Writes f32 and/or bf16 (either may be null).
// ---------------------------------------------------------------------------
__global__ __launch_bounds__(256) void ln_kernel(const float* __restrict__ X,
                                                 const float* __restrict__ g,
                                                 const float* __restrict__ b,
                                                 float* __restrict__ Yf,
                                                 __bf16* __restrict__ Yb) {
  __shared__ float red[256];
  int tid = threadIdx.x;
  const float* p = X + (size_t)blockIdx.x * MD;
  float v[4]; float s = 0.f;
#pragma unroll
  for (int i = 0; i < 4; ++i) { v[i] = p[i * 256 + tid]; s += v[i]; }
  red[tid] = s; __syncthreads();
  for (int st = 128; st > 0; st >>= 1) {
    if (tid < st) red[tid] += red[tid + st];
    __syncthreads();
  }
  float mu = red[0] * (1.f / MD); __syncthreads();
  float q = 0.f;
#pragma unroll
  for (int i = 0; i < 4; ++i) { float d = v[i] - mu; q += d * d; }
  red[tid] = q; __syncthreads();
  for (int st = 128; st > 0; st >>= 1) {
    if (tid < st) red[tid] += red[tid + st];
    __syncthreads();
  }
  float inv = rsqrtf(red[0] * (1.f / MD) + 1e-6f);
#pragma unroll
  for (int i = 0; i < 4; ++i) {
    int c = i * 256 + tid;
    float o = (v[i] - mu) * inv * g[c] + b[c];
    if (Yf) Yf[(size_t)blockIdx.x * MD + c] = o;
    if (Yb) Yb[(size_t)blockIdx.x * MD + c] = f2bf(o);
  }
}

// out[row] = sigmoid(dot(Y[row], kd_w) + kd_b)
__global__ __launch_bounds__(256) void kd_kernel(const float* __restrict__ Y,
                                                 const float* __restrict__ w,
                                                 const float* __restrict__ b,
                                                 float* __restrict__ out) {
  __shared__ float red[256];
  int tid = threadIdx.x;
  const float* p = Y + (size_t)blockIdx.x * MD;
  float s = 0.f;
#pragma unroll
  for (int i = 0; i < 4; ++i) { int c = i * 256 + tid; s += p[c] * w[c]; }
  red[tid] = s; __syncthreads();
  for (int st = 128; st > 0; st >>= 1) {
    if (tid < st) red[tid] += red[tid + st];
    __syncthreads();
  }
  if (tid == 0) {
    float z = red[0] + b[0];
    out[blockIdx.x] = 1.f / (1.f + __expf(-z));
  }
}

// ---------------------------------------------------------------------------
extern "C" void kernel_launch(void* const* d_in, const int* in_sizes, int n_in,
                              void* d_out, int out_size, void* d_ws, size_t ws_size,
                              hipStream_t stream) {
  (void)in_sizes; (void)n_in; (void)out_size; (void)ws_size;
  const float* x    = (const float*)d_in[0];
  const float* Wq   = (const float*)d_in[2];
  const float* Wk   = (const float*)d_in[3];
  const float* Wv   = (const float*)d_in[4];
  const float* Wo   = (const float*)d_in[5];
  const float* fc1w = (const float*)d_in[6];
  const float* fc1b = (const float*)d_in[7];
  const float* kaw  = (const float*)d_in[8];
  const float* kab  = (const float*)d_in[9];
  const float* kdw  = (const float*)d_in[10];
  const float* kdb  = (const float*)d_in[11];
  const float* lnyg = (const float*)d_in[12];
  const float* lnyb = (const float*)d_in[13];
  const float* lnkg = (const float*)d_in[14];
  const float* lnkb = (const float*)d_in[15];

  float* out = (float*)d_out;      // 8192 sigmoid outputs
  float* aw  = out + NR;           // 8192x8192 f32 attention weights (output 2)

  // ---- workspace layout (byte offsets; peak ~284 MB, regions reused) ----
  char* w = (char*)d_ws;
  const size_t MB = 1024 * 1024;
  __bf16* xb    = (__bf16*)(w + 0 * MB);     // 16 MB
  __bf16* wqb   = (__bf16*)(w + 16 * MB);    //  2 MB
  __bf16* wkb   = (__bf16*)(w + 18 * MB);
  __bf16* wvb   = (__bf16*)(w + 20 * MB);
  __bf16* wob   = (__bf16*)(w + 22 * MB);
  __bf16* fc1wb = (__bf16*)(w + 24 * MB);
  __bf16* kawb  = (__bf16*)(w + 26 * MB);
  __bf16* db1   = (__bf16*)(w + 28 * MB);    // 16 MB each
  __bf16* db2   = (__bf16*)(w + 44 * MB);
  __bf16* db3   = (__bf16*)(w + 60 * MB);
  __bf16* Qb    = (__bf16*)(w + 76 * MB);    // 16 MB
  __bf16* Kb    = (__bf16*)(w + 92 * MB);    // 16 MB
  __bf16* Vt    = (__bf16*)(w + 108 * MB);   // 16 MB (transposed MD x NR)
  float*  MN    = (float*)(w + 124 * MB);    // 32 MB
  __bf16* awT   = (__bf16*)(w + 156 * MB);   // 128 MB (transposed bf16 aw)
  __bf16* Yb    = (__bf16*)(w + 28 * MB);    // reuse db1 (dead after mn)
  float*  Y2    = (float*)(w + 44 * MB);     // reuse db2+db3
  __bf16* Y3b   = (__bf16*)(w + 76 * MB);    // reuse Qb (dead after logits)
  float*  Y4    = (float*)(w + 92 * MB);     // reuse Kb+Vt (dead after avt)
  float*  Y5    = (float*)(w + 124 * MB);    // reuse MN (dead after Wo)

  dim3 blk(256);
  dim3 gN(MD / 256, NR / 128);    // (4, 64) for 1024-wide GEMM outputs
  dim3 gL(NR / 256, NR / 128);    // (32, 64) for 8192-wide logits

  // ---- stage bf16 operands (one-shot streaming conversions) ----
  cvt_bf16_kernel<<<NR * MD / 256, blk, 0, stream>>>(x, xb, NR * MD);
  cvt_bf16_kernel<<<MD * MD / 256, blk, 0, stream>>>(Wq, wqb, MD * MD);
  cvt_bf16_kernel<<<MD * MD / 256, blk, 0, stream>>>(Wk, wkb, MD * MD);
  cvt_bf16_kernel<<<MD * MD / 256, blk, 0, stream>>>(Wv, wvb, MD * MD);
  cvt_bf16_kernel<<<MD * MD / 256, blk, 0, stream>>>(Wo, wob, MD * MD);
  cvt_bf16_kernel<<<MD * MD / 256, blk, 0, stream>>>(fc1w, fc1wb, MD * MD);
  cvt_bf16_kernel<<<MD * MD / 256, blk, 0, stream>>>(kaw, kawb, MD * MD);
  diff_kernel<<<NR * MD / 256, blk, 0, stream>>>(x, db1, db2, db3);

  // ---- projections: Q (x0.06) and K -> bf16 row major; V -> bf16 transposed
  gemm_bf16_kernel<<<gN, blk, 0, stream>>>(xb, wqb, MD, MD, nullptr, Qb, nullptr, 0,
                                           0.06f, nullptr, nullptr, 0);
  gemm_bf16_kernel<<<gN, blk, 0, stream>>>(xb, wkb, MD, MD, nullptr, Kb, nullptr, 0,
                                           1.0f, nullptr, nullptr, 0);
  gemm_bf16_kernel<<<gN, blk, 0, stream>>>(xb, wvb, MD, MD, nullptr, nullptr, Vt, NR,
                                           1.0f, nullptr, nullptr, 0);

  // ---- mn branch: fused 3-diff GEMM + relu-sum; tail rows copied from x ----
  mn_gemm_kernel<<<dim3(MD / 128, NR / 64), blk, 0, stream>>>(db1, db2, db3,
                                                              fc1wb, fc1b, MN);
  mn_tail_kernel<<<(4 * MD + 255) / 256, blk, 0, stream>>>(x, MN);

  // ---- logits straight into the aw output region, softmax in place ----
  gemm_bf16_kernel<<<gL, blk, 0, stream>>>(Qb, Kb, MD, NR, aw, nullptr, nullptr, 0,
                                           1.0f, nullptr, nullptr, 0);
  softmax_kernel<<<NR, blk, 0, stream>>>(aw);

  // ---- y = aw^T @ V: stage awT bf16 via LDS transpose, then K-contiguous GEMM
  transpose_bf_kernel<<<dim3(NR / 64, NR / 64), blk, 0, stream>>>(aw, awT);
  gemm_bf16_kernel<<<gN, blk, 0, stream>>>(awT, Vt, NR, MD, nullptr, Yb, nullptr, 0,
                                           1.0f, nullptr, nullptr, 0);

  // ---- Wo projection fused with +mn residual -> f32 ----
  gemm_bf16_kernel<<<gN, blk, 0, stream>>>(Yb, wob, MD, MD, Y2, nullptr, nullptr, 0,
                                           1.0f, nullptr, MN, 0);

  // ---- LN -> relu(ka) -> LN -> sigmoid(kd) ----
  ln_kernel<<<NR, blk, 0, stream>>>(Y2, lnyg, lnyb, nullptr, Y3b);
  gemm_bf16_kernel<<<gN, blk, 0, stream>>>(Y3b, kawb, MD, MD, Y4, nullptr, nullptr, 0,
                                           1.0f, kab, nullptr, 1);
  ln_kernel<<<NR, blk, 0, stream>>>(Y4, lnkg, lnkb, Y5, nullptr);
  kd_kernel<<<NR, blk, 0, stream>>>(Y5, kdw, kdb, out);
}
